// MultiheadSpatialBlock_37838661878170
// MI455X (gfx1250) — matrix operand order repaired and checked by reading, not compiled, not run on hardware
//
#include <hip/hip_runtime.h>

typedef __attribute__((ext_vector_type(16))) _Float16 v16h;
typedef __attribute__((ext_vector_type(8)))  _Float16 v8h;
typedef __attribute__((ext_vector_type(8)))  float    v8f;

#define B_     16
#define C_     256
#define HW_    9216
#define PL_    64
#define HEADS_ 4
#define G_     2304
#define EPS_   1e-5f

#define NT_    128        // spatial columns per GEMM block
#define KP_    264        // padded K stride (halves) in LDS
#define HP_    72         // padded plane stride for H staging in LDS

// workspace layout (bytes)
#define W1H_OFF 0                      // 64*256 f16   = 32768
#define SC_OFF  32768                  // 64 f32
#define SH_OFF  33024                  // 64 f32
#define SWM_OFF 33280                  // 4  f32
#define U_OFF   33536                  // 16*4*64 f32  = 16384
#define V_OFF   49920                  // 16*4*64 f32  = 16384
#define D_OFF   66304                  // 16*4 f32
#define H_OFF   66560                  // 16*9216*64 f16 = 18874368

union V16 { v16h v; v8h h[2]; };

// ---------------- kernel 1: fold BN, convert W1 to f16, sum(wm) per head ----
__global__ __launch_bounds__(256) void k_prep(const float* __restrict__ w1,
    const float* __restrict__ b1, const float* __restrict__ gamma,
    const float* __restrict__ beta, const float* __restrict__ mean,
    const float* __restrict__ var, const float* __restrict__ wm, char* ws) {
  _Float16* w1h = (_Float16*)(ws + W1H_OFF);
  float* sc  = (float*)(ws + SC_OFF);
  float* sh  = (float*)(ws + SH_OFF);
  float* swm = (float*)(ws + SWM_OFF);
  int t = threadIdx.x;
  for (int i = t; i < PL_ * C_; i += 256) w1h[i] = (_Float16)w1[i];
  if (t < PL_) {
    float s = gamma[t] * rsqrtf(var[t] + EPS_);
    sc[t] = s;
    sh[t] = (b1[t] - mean[t]) * s + beta[t];   // GEMM out * sc + sh, then ReLU
  }
  __shared__ float part[8][HEADS_];
  int lane = t & 31, wid = t >> 5;
  for (int h = 0; h < HEADS_; ++h) {
    float p = 0.f;
    for (int g = t; g < G_; g += 256) p += wm[h * G_ + g];
    for (int off = 16; off; off >>= 1) p += __shfl_xor(p, off, 32);
    if (lane == 0) part[wid][h] = p;
  }
  __syncthreads();
  if (t < HEADS_) {
    float s = 0.f;
    for (int w = 0; w < 8; ++w) s += part[w][t];
    swm[t] = s;
  }
}

// ---------------- kernel 2: H = relu(BN(W1 @ X)) via WMMA, store f16 [b][s][64]
__global__ __launch_bounds__(256) void k_gemm1(const float* __restrict__ x,
                                               char* ws) {
  __shared__ _Float16 lds[NT_ * KP_ + PL_ * KP_];
  _Float16* Xs  = lds;               // [NT_][KP_]  (X tile transposed, f16)
  _Float16* W1s = lds + NT_ * KP_;   // [PL_][KP_]
  const _Float16* w1h = (const _Float16*)(ws + W1H_OFF);
  const float* sc = (const float*)(ws + SC_OFF);
  const float* sh = (const float*)(ws + SH_OFF);
  _Float16* Hg = (_Float16*)(ws + H_OFF);

  int t  = threadIdx.x;
  int b  = blockIdx.x / (HW_ / NT_);
  int s0 = (blockIdx.x % (HW_ / NT_)) * NT_;

  // W1 (f16, [64][256]) -> LDS, 2048 x uint4
  {
    const uint4* src = (const uint4*)w1h;
    for (int i = 0; i < 8; ++i) {
      int g = i * 256 + t;
      int row = g >> 5;
      int c8  = (g & 31) << 3;
      *(uint4*)&W1s[row * KP_ + c8] = src[g];
    }
  }
  // X tile [c][s0..s0+127] fp32 -> f16 transposed Xs[s][c]
  for (int i = 0; i < 32; ++i) {
    int g  = i * 256 + t;
    int c  = g >> 5;
    int s4 = (g & 31) << 2;
    const float4 f = *(const float4*)(x + ((size_t)(b * C_ + c)) * HW_ + s0 + s4);
    Xs[(s4 + 0) * KP_ + c] = (_Float16)f.x;
    Xs[(s4 + 1) * KP_ + c] = (_Float16)f.y;
    Xs[(s4 + 2) * KP_ + c] = (_Float16)f.z;
    Xs[(s4 + 3) * KP_ + c] = (_Float16)f.w;
  }
  __syncthreads();

  int lane = t & 31, wave = t >> 5;
  int pt = wave & 3;         // plane tile (16 planes)
  int cw = wave >> 1 & 0;    // unused placeholder
  (void)cw;
  int cg = wave >> 2;        // 0..1 : owns col tiles cg*4 .. cg*4+3
  int lr = lane & 15;
  int lh = lane >> 4;

  v8f acc[4] = {};
  int arow = (pt * 16 + lr) * KP_;
  for (int kk = 0; kk < 8; ++kk) {
    int kbA = kk * 32 + lh * 8;        // A: 16-bit layout, K {0..7,16..23}/{8..15,24..31}
    V16 a;
    a.h[0] = *(const v8h*)&W1s[arow + kbA];
    a.h[1] = *(const v8h*)&W1s[arow + kbA + 16];
    int kbB = kk * 32 + lh * 8;        // B: 16-bit layout, K {0..7,16..23}/{8..15,24..31}
#pragma unroll
    for (int j = 0; j < 4; ++j) {
      int s = (cg * 4 + j) * 16 + lr;
      V16 bb;
      bb.h[0] = *(const v8h*)&Xs[s * KP_ + kbB];
      bb.h[1] = *(const v8h*)&Xs[s * KP_ + kbB + 16];
      acc[j] = __builtin_amdgcn_wmma_f32_16x16x32_f16(false, a.v, false, bb.v,
                                                      (short)0, acc[j], false, false);
    }
  }
  __syncthreads();                     // all reads of Xs done; reuse as H staging
  _Float16* Hs = Xs;                   // [NT_][HP_]
#pragma unroll
  for (int j = 0; j < 4; ++j) {
    int s = (cg * 4 + j) * 16 + lr;
#pragma unroll
    for (int r = 0; r < 8; ++r) {
      int p = pt * 16 + lh * 8 + r;    // D layout: VGPR r -> M = r + 8*(lane/16)
      float vv = acc[j][r] * sc[p] + sh[p];
      vv = vv > 0.f ? vv : 0.f;
      Hs[s * HP_ + p] = (_Float16)vv;
    }
  }
  __syncthreads();
  // coalesced store: H[b][s0+s][0..63] f16
  for (int i = 0; i < 4; ++i) {
    int g  = i * 256 + t;
    int s  = g >> 3;
    int c8 = (g & 7) << 3;
    uint4 q = *(const uint4*)&Hs[s * HP_ + c8];
    *(uint4*)&Hg[((size_t)(b * HW_) + s0 + s) * PL_ + c8] = q;
  }
}

// ---------------- kernel 3: u[b,h,p] = sum_g H[b, h*G+g, p] * wm[h,g] --------
__global__ __launch_bounds__(256) void k_u(const float* __restrict__ wm, char* ws) {
  const _Float16* Hg = (const _Float16*)(ws + H_OFF);
  float* ug = (float*)(ws + U_OFF);
  int b = blockIdx.x >> 2, head = blockIdx.x & 3;
  int t = threadIdx.x, lane = t & 31, wid = t >> 5;
  float acc[PL_];
#pragma unroll
  for (int p = 0; p < PL_; ++p) acc[p] = 0.f;
  for (int s = t; s < G_; s += 256) {
    float w = wm[head * G_ + s];
    const v8h* row = (const v8h*)(Hg + ((size_t)(b * HW_) + head * G_ + s) * PL_);
#pragma unroll
    for (int i = 0; i < 8; ++i) {
      v8h h = row[i];
#pragma unroll
      for (int j = 0; j < 8; ++j) acc[i * 8 + j] += w * (float)h[j];
    }
  }
  __shared__ float part[8][PL_];
#pragma unroll
  for (int p = 0; p < PL_; ++p) {
    float v = acc[p];
    for (int off = 16; off; off >>= 1) v += __shfl_xor(v, off, 32);
    if (lane == 0) part[wid][p] = v;
  }
  __syncthreads();
  if (t < PL_) {
    float s = 0.f;
    for (int w = 0; w < 8; ++w) s += part[w][t];
    ug[(b * HEADS_ + head) * PL_ + t] = s;
  }
}

// ---------------- kernel 4: logits -> softmax -> v = W2^T mask, d = mask.b2 --
__global__ __launch_bounds__(256) void k_mask(const float* __restrict__ w2,
    const float* __restrict__ b2, const float* __restrict__ bm, char* ws) {
  const float* ug  = (const float*)(ws + U_OFF);
  const float* swm = (const float*)(ws + SWM_OFF);
  float* vg = (float*)(ws + V_OFF);
  float* dg = (float*)(ws + D_OFF);
  int b = blockIdx.x >> 2, head = blockIdx.x & 3;
  int c = threadIdx.x, lane = c & 31, wid = c >> 5;
  __shared__ float us[PL_];
  __shared__ float red[8];
  __shared__ float vpart[8][PL_];
  if (c < PL_) us[c] = ug[(b * HEADS_ + head) * PL_ + c];
  __syncthreads();
  float wr[PL_];
#pragma unroll
  for (int p = 0; p < PL_; ++p) wr[p] = w2[c * PL_ + p];
  float lg = bm[head] + b2[c] * swm[head];
#pragma unroll
  for (int p = 0; p < PL_; ++p) lg += wr[p] * us[p];
  // block max
  float m = lg;
  for (int off = 16; off; off >>= 1) m = fmaxf(m, __shfl_xor(m, off, 32));
  if (lane == 0) red[wid] = m;
  __syncthreads();
  if (c == 0) { float mm = red[0]; for (int w = 1; w < 8; ++w) mm = fmaxf(mm, red[w]); red[0] = mm; }
  __syncthreads();
  float e = expf(lg - red[0]);
  __syncthreads();
  // block sum
  float ssum = e;
  for (int off = 16; off; off >>= 1) ssum += __shfl_xor(ssum, off, 32);
  if (lane == 0) red[wid] = ssum;
  __syncthreads();
  if (c == 0) { float s = 0.f; for (int w = 0; w < 8; ++w) s += red[w]; red[0] = s; }
  __syncthreads();
  float mask = e / red[0];
  // v[p] = sum_c mask * w2[c][p]
#pragma unroll
  for (int p = 0; p < PL_; ++p) {
    float v = mask * wr[p];
    for (int off = 16; off; off >>= 1) v += __shfl_xor(v, off, 32);
    if (lane == 0) vpart[wid][p] = v;
  }
  float dv = mask * b2[c];
  for (int off = 16; off; off >>= 1) dv += __shfl_xor(dv, off, 32);
  __syncthreads();                    // everyone has read red[0]
  if (lane == 0) red[wid] = dv;
  __syncthreads();
  if (c < PL_) {
    float s = 0.f;
    for (int w = 0; w < 8; ++w) s += vpart[w][c];
    vg[(b * HEADS_ + head) * PL_ + c] = s;
  }
  if (c == 0) {
    float s = 0.f;
    for (int w = 0; w < 8; ++w) s += red[w];
    dg[b * HEADS_ + head] = s;
  }
}

// ---------------- kernel 5: ctx[s] = v.H[:,s] + d ; out = x + ctx (broadcast C)
__global__ __launch_bounds__(256) void k_out(const float* __restrict__ x,
                                             float* __restrict__ out, char* ws) {
  const _Float16* Hg = (const _Float16*)(ws + H_OFF);
  const float* vg = (const float*)(ws + V_OFF);
  const float* dg = (const float*)(ws + D_OFF);
  __shared__ float vl[PL_];
  __shared__ float ctxs[256];
  int t = threadIdx.x;
  int b = blockIdx.x / 36;
  int chunk = blockIdx.x % 36;
  int s0 = chunk * 256;                // 256 | 2304, so one head per chunk
  int head = s0 / G_;
  if (t < PL_) vl[t] = vg[(b * HEADS_ + head) * PL_ + t];
  __syncthreads();
  {
    const v8h* row = (const v8h*)(Hg + ((size_t)(b * HW_) + s0 + t) * PL_);
    float s = dg[b * HEADS_ + head];
#pragma unroll
    for (int i = 0; i < 8; ++i) {
      v8h h = row[i];
#pragma unroll
      for (int j = 0; j < 8; ++j) s += vl[i * 8 + j] * (float)h[j];
    }
    ctxs[t] = s;
  }
  __syncthreads();
  int s4 = (t & 63) << 2;
  int cb = t >> 6;
  for (int cc = 0; cc < 64; ++cc) {
    int c = cc * 4 + cb;
    size_t idx = ((size_t)(b * C_ + c)) * HW_ + s0 + s4;
    float4 f = *(const float4*)(x + idx);
    f.x += ctxs[s4 + 0];
    f.y += ctxs[s4 + 1];
    f.z += ctxs[s4 + 2];
    f.w += ctxs[s4 + 3];
    *(float4*)(out + idx) = f;
  }
}

extern "C" void kernel_launch(void* const* d_in, const int* in_sizes, int n_in,
                              void* d_out, int out_size, void* d_ws, size_t ws_size,
                              hipStream_t stream) {
  (void)in_sizes; (void)n_in; (void)out_size; (void)ws_size;
  const float* x     = (const float*)d_in[0];
  const float* w1    = (const float*)d_in[1];
  const float* b1    = (const float*)d_in[2];
  const float* gamma = (const float*)d_in[3];
  const float* beta  = (const float*)d_in[4];
  const float* mean  = (const float*)d_in[5];
  const float* var   = (const float*)d_in[6];
  const float* w2    = (const float*)d_in[7];
  const float* b2    = (const float*)d_in[8];
  const float* wm    = (const float*)d_in[9];
  const float* bm    = (const float*)d_in[10];
  char* ws = (char*)d_ws;
  float* out = (float*)d_out;

  hipLaunchKernelGGL(k_prep, dim3(1), dim3(256), 0, stream,
                     w1, b1, gamma, beta, mean, var, wm, ws);
  hipLaunchKernelGGL(k_gemm1, dim3(B_ * (HW_ / NT_)), dim3(256), 0, stream, x, ws);
  hipLaunchKernelGGL(k_u, dim3(B_ * HEADS_), dim3(256), 0, stream, wm, ws);
  hipLaunchKernelGGL(k_mask, dim3(B_ * HEADS_), dim3(256), 0, stream, w2, b2, bm, ws);
  hipLaunchKernelGGL(k_out, dim3(B_ * 36), dim3(256), 0, stream, x, out, ws);
}
